// EncoderLayer_23811298689034
// MI455X (gfx1250) — matrix operand order repaired and checked by reading, not compiled, not run on hardware
//
#include <hip/hip_runtime.h>
#include <hip/hip_bf16.h>
#include <math.h>

typedef __bf16 bf16_t;
typedef bf16_t v8bf  __attribute__((ext_vector_type(8)));
typedef bf16_t v16bf __attribute__((ext_vector_type(16)));
typedef float  v8f   __attribute__((ext_vector_type(8)));
typedef unsigned int v4u __attribute__((ext_vector_type(4)));
typedef int v8i __attribute__((ext_vector_type(8)));
typedef int v4i __attribute__((ext_vector_type(4)));

#define D_MODEL 1024
#define FF_HIDDEN 4096
#define NUM_HEADS 16
#define HEAD_DIM 64
#define SEQ 2048
#define BATCH 2
#define MROWS (BATCH*SEQ)   /* 4096 */

#if __has_builtin(__builtin_amdgcn_tensor_load_to_lds)
#define HAVE_TDM 1
#else
#define HAVE_TDM 0
#endif

static __device__ __forceinline__ bf16_t f2bf(float f) {
  unsigned u = __float_as_uint(f);
  u += 0x7fffu + ((u >> 16) & 1u);              // round-to-nearest-even
  unsigned short h = (unsigned short)(u >> 16);
  union { unsigned short s; bf16_t b; } cv; cv.s = h;
  return cv.b;
}

static __device__ __forceinline__ v16bf cat16(v8bf lo, v8bf hi) {
  return __builtin_shufflevector(lo, hi, 0,1,2,3,4,5,6,7,8,9,10,11,12,13,14,15);
}

static __device__ __forceinline__ v8f wmma_bf16(v16bf a, v16bf b, v8f c) {
  return __builtin_amdgcn_wmma_f32_16x16x32_bf16(false, a, false, b, (short)0, c,
                                                 false, false);
}

#if HAVE_TDM
/* Issue one TDM 2D tile load (bf16 elements) global -> LDS.
   D# packing per CDNA5 ISA ch.8: group0 {count/type/lds/global addr},
   group1 {data_size, pad, tensor dims, tile dims, strides}.
   padIntCode: LDS pad every 2^(c+1) DWORDs; padAmtCode: pad (c+1) DWORDs.
   clang-23 builtin form: (v4u g0, v8i g1, v4i g2, v4i g3, v8i g4, i32 cpol). */
static __device__ __forceinline__ void tdm_load_2d_bf16(
    const bf16_t* gaddr, bf16_t* lds,
    unsigned tensor_d0, unsigned tensor_d1,
    unsigned tile_d0, unsigned tile_d1,
    unsigned long long stride0_elems,
    unsigned padIntCode, unsigned padAmtCode)
{
  unsigned long long ga = (unsigned long long)(uintptr_t)gaddr;
  unsigned ldsoff = (unsigned)(uintptr_t)lds;      // low 32 bits = LDS byte offset
  v4u g0;
  g0[0] = 1u;                                       // count=1, user descriptor
  g0[1] = ldsoff;                                   // lds_addr
  g0[2] = (unsigned)(ga & 0xffffffffu);             // global_addr[31:0]
  g0[3] = (unsigned)((ga >> 32) & 0x01ffffffu) | (2u << 30);  // addr[56:32] | type=2
  v8i g1;
  g1[0] = (int)((1u << 16) |                        // data_size = 2 bytes
                (1u << 20) |                        // pad_enable
                (padIntCode << 22) | (padAmtCode << 25));
  g1[1] = (int)((tensor_d0 & 0xffffu) << 16);       // tensor_dim0[15:0] (bits 63:48)
  g1[2] = (int)((tensor_d0 >> 16) | ((tensor_d1 & 0xffffu) << 16));
  g1[3] = (int)((tensor_d1 >> 16) | (tile_d0 << 16));
  g1[4] = (int)(tile_d1 & 0xffffu);                 // tile_dim1; tile_dim2 = 0
  g1[5] = (int)(stride0_elems & 0xffffffffu);       // tensor_dim0_stride[31:0]
  g1[6] = (int)((stride0_elems >> 32) & 0xffffu);   // stride[47:32]; dim1_stride=0
  g1[7] = 0;
  v4i z4 = {0, 0, 0, 0};
  v8i z8 = {0, 0, 0, 0, 0, 0, 0, 0};
  __builtin_amdgcn_tensor_load_to_lds(g0, g1, z4, z4, z8, 0);
}
#endif

/* ---------------- fp32 -> bf16 elementwise ---------------- */
__global__ void cvt_f32_bf16(const float* __restrict__ in, bf16_t* __restrict__ out, int n) {
  int i = blockIdx.x * 256 + threadIdx.x;
  if (i < n) out[i] = f2bf(in[i]);
}

/* ------------- transpose fp32 [K][N] -> bf16 [N][K] ------------- */
__global__ void transp_bf16(const float* __restrict__ W, bf16_t* __restrict__ WT,
                            int K, int N) {
  __shared__ float t[32][33];
  int n0 = blockIdx.x * 32, k0 = blockIdx.y * 32;
  for (int i = threadIdx.y; i < 32; i += 8)
    t[i][threadIdx.x] = W[(size_t)(k0 + i) * N + n0 + threadIdx.x];
  __syncthreads();
  for (int i = threadIdx.y; i < 32; i += 8)
    WT[(size_t)(n0 + i) * K + k0 + threadIdx.x] = f2bf(t[threadIdx.x][i]);
}

/* ---------------- bf16 WMMA GEMM: out = A[M,K] * BT[N,K]^T + bias ----------------
   Block = 256 threads (8 waves), tile 128x128, k-step 32.
   TDM double-buffered staging: wave 0 issues tensor_load_to_lds for tile i+1
   while all 8 waves run WMMAs on tile i.                                        */
template<bool RELU, bool BF16OUT>
__global__ __launch_bounds__(256)
void gemm_bf16_wmma(const bf16_t* __restrict__ A, const bf16_t* __restrict__ BT,
                    const float* __restrict__ bias,
                    float* __restrict__ outF, bf16_t* __restrict__ outB,
                    int M, int N, int K) {
#if HAVE_TDM
  __shared__ bf16_t Alds[2][128 * 40];
  __shared__ bf16_t Blds[2][128 * 40];
#else
  __shared__ bf16_t Alds[1][128 * 40];
  __shared__ bf16_t Blds[1][128 * 40];
#endif
  const int tid = threadIdx.x;
  const int lane = tid & 31, w = tid >> 5;
  const int half = lane >> 4, mrow = lane & 15;
  const int wm = w >> 2, wn = w & 3;
  const int m0 = blockIdx.y * 128, n0 = blockIdx.x * 128;

  v8f zero = {0.f,0.f,0.f,0.f,0.f,0.f,0.f,0.f};
  v8f acc[4][2];
#pragma unroll
  for (int i = 0; i < 4; i++)
#pragma unroll
    for (int j = 0; j < 2; j++) acc[i][j] = zero;

  const bf16_t* Abase = A + (size_t)m0 * K;
  const bf16_t* Bbase = BT + (size_t)n0 * K;
  const int nk = K >> 5;

#if HAVE_TDM
  if (w == 0) {   // prologue: fetch tile 0 (row 64B + 16B pad -> stride 40 elems)
    tdm_load_2d_bf16(Abase, &Alds[0][0], (unsigned)K, 128u, 32u, 128u,
                     (unsigned long long)K, 3u, 3u);
    tdm_load_2d_bf16(Bbase, &Blds[0][0], (unsigned)K, 128u, 32u, 128u,
                     (unsigned long long)K, 3u, 3u);
    __builtin_amdgcn_s_wait_tensorcnt(0);
  }
  __syncthreads();
#endif

  for (int it = 0; it < nk; ++it) {
#if HAVE_TDM
    const int cur = it & 1, nxt = cur ^ 1;
    if (w == 0 && it + 1 < nk) {   // async prefetch of next tile while computing
      tdm_load_2d_bf16(Abase + (size_t)(it + 1) * 32, &Alds[nxt][0],
                       (unsigned)K, 128u, 32u, 128u, (unsigned long long)K, 3u, 3u);
      tdm_load_2d_bf16(Bbase + (size_t)(it + 1) * 32, &Blds[nxt][0],
                       (unsigned)K, 128u, 32u, 128u, (unsigned long long)K, 3u, 3u);
    }
#else
    const int cur = 0;
    const int kt = it * 32;
#pragma unroll
    for (int i = 0; i < 2; i++) {                 // manual stage 128x32 of A and BT
      int idx = tid + i * 256;
      int row = idx >> 2, seg = idx & 3;
      *(v8bf*)(&Alds[0][row * 40 + seg * 8]) =
          *(const v8bf*)(Abase + (size_t)row * K + kt + seg * 8);
      *(v8bf*)(&Blds[0][row * 40 + seg * 8]) =
          *(const v8bf*)(Bbase + (size_t)row * K + kt + seg * 8);
    }
    __syncthreads();
#endif

    v16bf af[4], bfr[2];
#pragma unroll
    for (int mt = 0; mt < 4; mt++) {              // A frag: k = half*8.., 16+half*8..
      const bf16_t* ap = &Alds[cur][((4 * wm + mt) * 16 + mrow) * 40];
      af[mt] = cat16(*(const v8bf*)(ap + half * 8),
                     *(const v8bf*)(ap + 16 + half * 8));
    }
#pragma unroll
    for (int nt = 0; nt < 2; nt++) {              // B frag: k = half*8.., 16+half*8..
      const bf16_t* bp = &Blds[cur][((2 * wn + nt) * 16 + mrow) * 40];
      bfr[nt] = cat16(*(const v8bf*)(bp + half * 8), *(const v8bf*)(bp + 16 + half * 8));
    }
#pragma unroll
    for (int mt = 0; mt < 4; mt++)
#pragma unroll
      for (int nt = 0; nt < 2; nt++)
        acc[mt][nt] = wmma_bf16(af[mt], bfr[nt], acc[mt][nt]);

#if HAVE_TDM
    if (w == 0) __builtin_amdgcn_s_wait_tensorcnt(0);
#endif
    __syncthreads();
  }

#pragma unroll
  for (int mt = 0; mt < 4; mt++)
#pragma unroll
    for (int nt = 0; nt < 2; nt++) {
      int col = n0 + (2 * wn + nt) * 16 + mrow;
      float bv = bias[col];
#pragma unroll
      for (int r = 0; r < 8; r++) {               // C/D: M = r + 8*half, N = mrow
        int row = m0 + (4 * wm + mt) * 16 + r + 8 * half;
        float v = acc[mt][nt][r] + bv;
        if (RELU) v = fmaxf(v, 0.f);
        if (BF16OUT) outB[(size_t)row * N + col] = f2bf(v);
        else         outF[(size_t)row * N + col] = v;
      }
    }
}

/* ---------------- flash attention, bf16 WMMA, one wave per 16 q-rows ---------------- */
__global__ __launch_bounds__(256)
void attention_wmma(const bf16_t* __restrict__ Q, const bf16_t* __restrict__ Km,
                    const bf16_t* __restrict__ V, bf16_t* __restrict__ O) {
  __shared__ bf16_t Klds[32 * 72];      // K tile rows kj, cols d (pad 72)
  __shared__ bf16_t VTlds[64 * 40];     // V^T: rows d, cols kj (pad 40)
  __shared__ bf16_t Plds[8][16 * 40];   // per-wave P relayout buffer

  const int tid = threadIdx.x;
  const int lane = tid & 31, w = tid >> 5;
  const int half = lane >> 4, mrow = lane & 15;
  const int bh = blockIdx.x;
  const int qc = bh & 15, h = (bh >> 4) & 15, bb = bh >> 8;
  const int row0 = bb * SEQ + qc * 128;
  const int hoff = h * HEAD_DIM;
  const int krow0 = bb * SEQ;

  // Q fragments (d-chunks 0..31 and 32..63) for this wave's 16 rows
  const int qrow = row0 + w * 16 + mrow;
  const bf16_t* qp = Q + (size_t)qrow * D_MODEL + hoff;
  v16bf qa0 = cat16(*(const v8bf*)(qp + half * 8),      *(const v8bf*)(qp + 16 + half * 8));
  v16bf qa1 = cat16(*(const v8bf*)(qp + 32 + half * 8), *(const v8bf*)(qp + 48 + half * 8));

  v8f zero = {0.f,0.f,0.f,0.f,0.f,0.f,0.f,0.f};
  v8f acc[4];
  float mr[8], lr[8];
#pragma unroll
  for (int o = 0; o < 4; o++) acc[o] = zero;
#pragma unroll
  for (int r = 0; r < 8; r++) { mr[r] = -1e30f; lr[r] = 0.f; }

  const float scale = 0.125f;            // 1/sqrt(64)

  for (int kt = 0; kt < SEQ; kt += 32) {
    __syncthreads();
#if HAVE_TDM
    if (w == 0)                           // K tile via TDM (row 128B + 16B pad -> 72)
      tdm_load_2d_bf16(Km + (size_t)(krow0 + kt) * D_MODEL + hoff, Klds,
                       (unsigned)D_MODEL, 32u, 64u, 32u,
                       (unsigned long long)D_MODEL, 4u, 3u);
#endif
    {                                     // stage V^T (manual scatter transpose)
      int kj = tid >> 3, seg = tid & 7;
      const size_t gr = (size_t)(krow0 + kt + kj) * D_MODEL + hoff + seg * 8;
#if !HAVE_TDM
      *(v8bf*)(&Klds[kj * 72 + seg * 8]) = *(const v8bf*)(Km + gr);
#endif
      v8bf vv = *(const v8bf*)(V + gr);
#pragma unroll
      for (int e = 0; e < 8; e++)
        VTlds[(seg * 8 + e) * 40 + kj] = vv[e];
    }
#if HAVE_TDM
    if (w == 0) __builtin_amdgcn_s_wait_tensorcnt(0);
#endif
    __syncthreads();

    // scores: c0 = kj 0..15, c1 = kj 16..31  (B^T of scores is K itself)
    v8f c0 = zero, c1 = zero;
    {
      const bf16_t* k0p = &Klds[mrow * 72];
      const bf16_t* k1p = &Klds[(16 + mrow) * 72];
      v16bf kb00 = cat16(*(const v8bf*)(k0p + half * 8), *(const v8bf*)(k0p + 16 + half * 8));
      v16bf kb01 = cat16(*(const v8bf*)(k0p + 32 + half * 8), *(const v8bf*)(k0p + 48 + half * 8));
      v16bf kb10 = cat16(*(const v8bf*)(k1p + half * 8), *(const v8bf*)(k1p + 16 + half * 8));
      v16bf kb11 = cat16(*(const v8bf*)(k1p + 32 + half * 8), *(const v8bf*)(k1p + 48 + half * 8));
      c0 = wmma_bf16(qa0, kb00, c0); c0 = wmma_bf16(qa1, kb01, c0);
      c1 = wmma_bf16(qa0, kb10, c1); c1 = wmma_bf16(qa1, kb11, c1);
    }

    // online softmax over this 16x32 score tile
    float mx[8];
#pragma unroll
    for (int r = 0; r < 8; r++) {
      c0[r] *= scale; c1[r] *= scale;
      mx[r] = fmaxf(c0[r], c1[r]);
    }
#pragma unroll
    for (int msk = 1; msk < 16; msk <<= 1)
#pragma unroll
      for (int r = 0; r < 8; r++) mx[r] = fmaxf(mx[r], __shfl_xor(mx[r], msk, 32));

    float sm[8], cf[8];
#pragma unroll
    for (int r = 0; r < 8; r++) {
      float mn = fmaxf(mr[r], mx[r]);
      cf[r] = __expf(mr[r] - mn);
      mr[r] = mn;
      c0[r] = __expf(c0[r] - mn);
      c1[r] = __expf(c1[r] - mn);
      sm[r] = c0[r] + c1[r];
    }
#pragma unroll
    for (int msk = 1; msk < 16; msk <<= 1)
#pragma unroll
      for (int r = 0; r < 8; r++) sm[r] += __shfl_xor(sm[r], msk, 32);
#pragma unroll
    for (int r = 0; r < 8; r++) lr[r] = lr[r] * cf[r] + sm[r];
#pragma unroll
    for (int o = 0; o < 4; o++)
#pragma unroll
      for (int r = 0; r < 8; r++) acc[o][r] *= cf[r];

    // relayout P (C-layout -> A-fragment) via wave-private LDS
    bf16_t* pw = Plds[w];
#pragma unroll
    for (int r = 0; r < 8; r++) {
      pw[(r + 8 * half) * 40 + mrow]      = f2bf(c0[r]);
      pw[(r + 8 * half) * 40 + 16 + mrow] = f2bf(c1[r]);
    }
    asm volatile("s_wait_dscnt 0" ::: "memory");   // wave-private DS ordering

    const bf16_t* pr = pw + mrow * 40;
    v16bf pa = cat16(*(const v8bf*)(pr + half * 8), *(const v8bf*)(pr + 16 + half * 8));

#pragma unroll
    for (int o = 0; o < 4; o++) {                  // out d-tiles, B = V^T rows
      const bf16_t* vp = &VTlds[(o * 16 + mrow) * 40];
      v16bf vb = cat16(*(const v8bf*)(vp + half * 8), *(const v8bf*)(vp + 16 + half * 8));
      acc[o] = wmma_bf16(pa, vb, acc[o]);
    }
  }

#pragma unroll
  for (int o = 0; o < 4; o++) {
    int col = hoff + o * 16 + mrow;
#pragma unroll
    for (int r = 0; r < 8; r++) {
      int orow = row0 + w * 16 + r + 8 * half;
      O[(size_t)orow * D_MODEL + col] = f2bf(acc[o][r] / lr[r]);
    }
  }
}

/* ---------------- layernorm of (a + b), faithful (x-mean)/(std+eps)*g+b ---------------- */
__global__ __launch_bounds__(256)
void ln_kernel(const float* __restrict__ A, const float* __restrict__ B,
               const float* __restrict__ g, const float* __restrict__ be,
               float* __restrict__ outF, bf16_t* __restrict__ outB) {
  int row = blockIdx.x, tid = threadIdx.x;
  const float* a = A + (size_t)row * D_MODEL;
  const float* b = B + (size_t)row * D_MODEL;
  float v[4], s = 0.f, sq = 0.f;
#pragma unroll
  for (int j = 0; j < 4; j++) {
    float x = a[tid + j * 256] + b[tid + j * 256];
    v[j] = x; s += x; sq += x * x;
  }
  for (int m = 1; m < 32; m <<= 1) { s += __shfl_xor(s, m, 32); sq += __shfl_xor(sq, m, 32); }
  __shared__ float ss[8], ssq[8];
  if ((tid & 31) == 0) { ss[tid >> 5] = s; ssq[tid >> 5] = sq; }
  __syncthreads();
  s = 0.f; sq = 0.f;
#pragma unroll
  for (int wv = 0; wv < 8; wv++) { s += ss[wv]; sq += ssq[wv]; }
  float mean = s * (1.f / D_MODEL);
  float var  = sq * (1.f / D_MODEL) - mean * mean;
  float inv  = 1.f / (sqrtf(fmaxf(var, 0.f)) + 1e-5f);
#pragma unroll
  for (int j = 0; j < 4; j++) {
    int c = tid + j * 256;
    float o = (v[j] - mean) * inv * g[c] + be[c];
    outF[(size_t)row * D_MODEL + c] = o;
    if (outB) outB[(size_t)row * D_MODEL + c] = f2bf(o);
  }
}

/* ============================== host launch ============================== */
extern "C" void kernel_launch(void* const* d_in, const int* in_sizes, int n_in,
                              void* d_out, int out_size, void* d_ws, size_t ws_size,
                              hipStream_t stream) {
  (void)in_sizes; (void)n_in; (void)out_size; (void)ws_size;
  const float* x  = (const float*)d_in[0];
  const float* Wq = (const float*)d_in[1];  const float* bq  = (const float*)d_in[2];
  const float* Wk = (const float*)d_in[3];  const float* bk  = (const float*)d_in[4];
  const float* Wv = (const float*)d_in[5];  const float* bv  = (const float*)d_in[6];
  const float* Wo = (const float*)d_in[7];  const float* bo  = (const float*)d_in[8];
  const float* W1 = (const float*)d_in[9];  const float* b1  = (const float*)d_in[10];
  const float* W2 = (const float*)d_in[11]; const float* b2  = (const float*)d_in[12];
  const float* g1 = (const float*)d_in[13]; const float* be1 = (const float*)d_in[14];
  const float* g2 = (const float*)d_in[15]; const float* be2 = (const float*)d_in[16];
  float* out = (float*)d_out;

  char* p = (char*)d_ws;
  auto alloc = [&](size_t bytes) -> char* {
    char* r = p; p += (bytes + 255) & ~(size_t)255; return r;
  };
  const size_t MD = (size_t)MROWS * D_MODEL;
  bf16_t* xb   = (bf16_t*)alloc(MD * 2);
  bf16_t* WqT  = (bf16_t*)alloc((size_t)D_MODEL * D_MODEL * 2);
  bf16_t* WkT  = (bf16_t*)alloc((size_t)D_MODEL * D_MODEL * 2);
  bf16_t* WvT  = (bf16_t*)alloc((size_t)D_MODEL * D_MODEL * 2);
  bf16_t* WoT  = (bf16_t*)alloc((size_t)D_MODEL * D_MODEL * 2);
  bf16_t* W1T  = (bf16_t*)alloc((size_t)FF_HIDDEN * D_MODEL * 2);
  bf16_t* W2T  = (bf16_t*)alloc((size_t)D_MODEL * FF_HIDDEN * 2);
  bf16_t* Qb   = (bf16_t*)alloc(MD * 2);
  bf16_t* Kb   = (bf16_t*)alloc(MD * 2);
  bf16_t* Vb   = (bf16_t*)alloc(MD * 2);
  bf16_t* attnb= (bf16_t*)alloc(MD * 2);
  float*  aout = (float*)alloc(MD * 4);
  float*  hbuf = (float*)alloc(MD * 4);
  bf16_t* hb   = (bf16_t*)alloc(MD * 2);
  bf16_t* ub   = (bf16_t*)alloc((size_t)MROWS * FF_HIDDEN * 2);
  float*  ff   = (float*)alloc(MD * 4);

  { int n = (int)MD; cvt_f32_bf16<<<(n + 255) / 256, 256, 0, stream>>>(x, xb, n); }

  dim3 tb(32, 8);
  transp_bf16<<<dim3(D_MODEL/32, D_MODEL/32), tb, 0, stream>>>(Wq, WqT, D_MODEL, D_MODEL);
  transp_bf16<<<dim3(D_MODEL/32, D_MODEL/32), tb, 0, stream>>>(Wk, WkT, D_MODEL, D_MODEL);
  transp_bf16<<<dim3(D_MODEL/32, D_MODEL/32), tb, 0, stream>>>(Wv, WvT, D_MODEL, D_MODEL);
  transp_bf16<<<dim3(D_MODEL/32, D_MODEL/32), tb, 0, stream>>>(Wo, WoT, D_MODEL, D_MODEL);
  transp_bf16<<<dim3(FF_HIDDEN/32, D_MODEL/32), tb, 0, stream>>>(W1, W1T, D_MODEL, FF_HIDDEN);
  transp_bf16<<<dim3(D_MODEL/32, FF_HIDDEN/32), tb, 0, stream>>>(W2, W2T, FF_HIDDEN, D_MODEL);

  dim3 gD(D_MODEL / 128, MROWS / 128);            // (8, 32)
  gemm_bf16_wmma<false, true ><<<gD, 256, 0, stream>>>(xb, WqT, bq, nullptr, Qb, MROWS, D_MODEL, D_MODEL);
  gemm_bf16_wmma<false, true ><<<gD, 256, 0, stream>>>(xb, WkT, bk, nullptr, Kb, MROWS, D_MODEL, D_MODEL);
  gemm_bf16_wmma<false, true ><<<gD, 256, 0, stream>>>(xb, WvT, bv, nullptr, Vb, MROWS, D_MODEL, D_MODEL);

  attention_wmma<<<BATCH * NUM_HEADS * (SEQ / 128), 256, 0, stream>>>(Qb, Kb, Vb, attnb);

  gemm_bf16_wmma<false, false><<<gD, 256, 0, stream>>>(attnb, WoT, bo, aout, nullptr, MROWS, D_MODEL, D_MODEL);

  ln_kernel<<<MROWS, 256, 0, stream>>>(x, aout, g1, be1, hbuf, hb);

  dim3 gF(FF_HIDDEN / 128, MROWS / 128);          // (32, 32)
  gemm_bf16_wmma<true,  true ><<<gF, 256, 0, stream>>>(hb, W1T, b1, nullptr, ub, MROWS, FF_HIDDEN, D_MODEL);
  gemm_bf16_wmma<false, false><<<gD, 256, 0, stream>>>(ub, W2T, b2, ff, nullptr, MROWS, D_MODEL, FF_HIDDEN);

  ln_kernel<<<MROWS, 256, 0, stream>>>(hbuf, ff, g2, be2, out, nullptr);
}
